// Answer2Cone_49572512530723
// MI455X (gfx1250) — matrix operand order repaired and checked by reading, not compiled, not run on hardware
//
#include <hip/hip_runtime.h>
#include <math.h>

#define N_NODES 50000
#define N_EDGES 600000
#define HDIM    128
#define N_GRAPHS 512
#define BN_EPS  1e-5f
#define SLOPE   0.2f
#define PI_F    3.14159265358979323846f

typedef __attribute__((ext_vector_type(16))) _Float16 v16h;
typedef __attribute__((ext_vector_type(8)))  float    v8f;

union Frag { v16h h; unsigned int u[8]; };

__device__ inline float atomicMaxF(float* addr, float val) {
  int* ia = (int*)addr;
  int old = __float_as_int(*addr);
  while (__int_as_float(old) < val) {
    int assumed = old;
    old = atomicCAS(ia, assumed, __float_as_int(val));
    if (old == assumed) break;
  }
  return __int_as_float(old);
}

__global__ void k_fill(float* p, float v, long n) {
  long i = (long)blockIdx.x * blockDim.x + threadIdx.x;
  if (i < n) p[i] = v;
}

// C[M,128] = act(A[M,128] @ W[128,128] + bias), act: 0=none 1=tanh 2=tanh*pi
// 8 waves/block; wave w owns 16-col tile w; 4 row-tiles of 16 per block.
// W pre-packed in LDS into exact B-fragment dwords (32B contiguous per lane
// per K-step -> 2x ds_load_b128); A slab 64x128 f16 (pairs contiguous -> b128).
// ks-outer / rt-inner with 4 live accumulators reuses each B fragment 4x.
__global__ __launch_bounds__(256) void k_gemm128(const float* __restrict__ A,
                                                 const float* __restrict__ W,
                                                 const float* __restrict__ bias,
                                                 float* __restrict__ C,
                                                 int M, int act) {
  __shared__ unsigned int ldsWB[8192];   // 32KB: [ks][wave][lane][v] packed pairs
  __shared__ _Float16   ldsA[64 * 128];  // 16KB
  int tid = threadIdx.x;

  // Pre-pack W[k][n] into B-fragment layout:
  // dword p = ((ks*8 + w)*32 + lane)*8 + v holds (W[k][n], W[k+1][n]) as 2xf16
  // with k = ks*32 + 2v + (lane>>4)*16, n = w*16 + (lane&15).
  for (int p = tid; p < 8192; p += 256) {
    int v = p & 7, lane = (p >> 3) & 31, w = (p >> 8) & 7, ks = p >> 11;
    int k = ks * 32 + (v << 1) + ((lane >> 4) << 4);
    int n = (w << 4) + (lane & 15);
    _Float16 lo = (_Float16)W[k * 128 + n];
    _Float16 hi = (_Float16)W[(k + 1) * 128 + n];
    unsigned int ulo = *(const unsigned short*)&lo;
    unsigned int uhi = *(const unsigned short*)&hi;
    ldsWB[p] = ulo | (uhi << 16);
  }
  long row0 = (long)blockIdx.x * 64;
  for (int i = tid; i < 64 * 128; i += 256) {
    long r = row0 + (i >> 7);
    ldsA[i] = (r < M) ? (_Float16)A[r * 128 + (i & 127)] : (_Float16)0.0f;
  }
  __syncthreads();

  int wv = tid >> 5, lane = tid & 31, half = lane >> 4, l16 = lane & 15;
  int col = (wv << 4) + l16;
  float bval = bias ? bias[col] : 0.0f;

  v8f acc[4];
#pragma unroll
  for (int rt = 0; rt < 4; ++rt) acc[rt] = (v8f){0.f,0.f,0.f,0.f,0.f,0.f,0.f,0.f};

#pragma unroll
  for (int ks = 0; ks < 4; ++ks) {
    Frag fb;
    const unsigned int* bp = &ldsWB[(((ks << 3) + wv) * 32 + lane) * 8];
#pragma unroll
    for (int v = 0; v < 8; ++v) fb.u[v] = bp[v];   // 2x ds_load_b128
    int kk = ks * 32;
#pragma unroll
    for (int rt = 0; rt < 4; ++rt) {
      Frag fa;
      const _Float16* Ar = &ldsA[(rt * 16 + l16) * 128];
#pragma unroll
      for (int v = 0; v < 8; ++v) {
        // A 16x32 f16 layout: VGPR v holds K = base, base+1 packed (contiguous)
        int k = kk + (v << 1) + (half << 4);
        fa.u[v] = *(const unsigned int*)(&Ar[k]); // merges to 2x ds_load_b128
      }
      acc[rt] = __builtin_amdgcn_wmma_f32_16x16x32_f16(false, fa.h, false, fb.h,
                                                       (short)0, acc[rt], false, false);
    }
  }

#pragma unroll
  for (int rt = 0; rt < 4; ++rt) {
#pragma unroll
    for (int r = 0; r < 8; ++r) {
      long row = row0 + rt * 16 + r + (half << 3);
      if (row < M) {
        float v = acc[rt][r] + bval;
        if (act == 1) v = tanhf(v);
        else if (act == 2) v = tanhf(v) * PI_F;
        C[row * 128 + col] = v;
      }
    }
  }
}

// wave per edge: loopEW[dst] += EW[e]; deg[dst] += 1
__global__ void k_scatter_loop(const float* __restrict__ EW, const int* __restrict__ dst,
                               float* deg, float* loopEW) {
  int gi = blockIdx.x * blockDim.x + threadIdx.x;
  int e = gi >> 5, lane = gi & 31;
  if (e >= N_EDGES) return;
  int d = dst[e];
#pragma unroll
  for (int j = 0; j < 4; ++j) {
    int c = lane * 4 + j;
    atomicAdd(&loopEW[(long)d * 128 + c], EW[(long)e * 128 + c]);
  }
  if (lane == 0) atomicAdd(&deg[d], 1.0f);
}

__global__ void k_loop_div(float* loopEW, const float* deg) {
  long i = (long)blockIdx.x * blockDim.x + threadIdx.x;
  if (i < (long)N_NODES * 128) loopEW[i] /= fmaxf(deg[i >> 7], 1.0f);
}

// wave per (edge or self-loop): score = leaky(xl[s]+xr[d]+ew) . att; atomicMax emax[d]
__global__ void k_score(const float* __restrict__ xl, const float* __restrict__ xr,
                        const float* __restrict__ EW, const float* __restrict__ loopEW,
                        const int* __restrict__ src, const int* __restrict__ dst,
                        const float* __restrict__ att, float* score, float* emax) {
  int gi = blockIdx.x * blockDim.x + threadIdx.x;
  int i = gi >> 5, lane = gi & 31;
  if (i >= N_EDGES + N_NODES) return;
  int s, d; const float* ew;
  if (i < N_EDGES) { s = src[i]; d = dst[i]; ew = EW + (long)i * 128; }
  else { s = d = i - N_EDGES; ew = loopEW + (long)s * 128; }
  float p = 0.f;
#pragma unroll
  for (int j = 0; j < 4; ++j) {
    int c = lane * 4 + j;
    float m = xl[(long)s * 128 + c] + xr[(long)d * 128 + c] + ew[c];
    m = (m > 0.f) ? m : SLOPE * m;
    p += m * att[c];
  }
#pragma unroll
  for (int off = 16; off > 0; off >>= 1) p += __shfl_xor(p, off, 32);
  if (lane == 0) { score[i] = p; atomicMaxF(&emax[d], p); }
}

__global__ void k_exden(const float* score, const float* emax, const int* dst,
                        float* ex, float* den) {
  int i = blockIdx.x * blockDim.x + threadIdx.x;
  if (i >= N_EDGES + N_NODES) return;
  int d = (i < N_EDGES) ? dst[i] : (i - N_EDGES);
  float e = expf(score[i] - emax[d]);
  ex[i] = e;
  atomicAdd(&den[d], e);
}

// wave per (edge/self-loop): out[d] += (ex/den[d]) * xl[s]
__global__ void k_aggr(const float* __restrict__ ex, const float* __restrict__ den,
                       const float* __restrict__ xl,
                       const int* __restrict__ src, const int* __restrict__ dst,
                       float* out) {
  int gi = blockIdx.x * blockDim.x + threadIdx.x;
  int i = gi >> 5, lane = gi & 31;
  if (i >= N_EDGES + N_NODES) return;
  int s, d;
  if (i < N_EDGES) { s = src[i]; d = dst[i]; } else { s = d = i - N_EDGES; }
  float alpha = ex[i] / den[d];
#pragma unroll
  for (int j = 0; j < 4; ++j) {
    int c = lane * 4 + j;
    atomicAdd(&out[(long)d * 128 + c], alpha * xl[(long)s * 128 + c]);
  }
}

// per-column sum / sum-of-squares, block-local reduce then 128 atomics/block
__global__ __launch_bounds__(256) void k_bnstats(const float* __restrict__ x,
                                                 float* sum, float* sq, int M) {
  int c = threadIdx.x & 127, sub = threadIdx.x >> 7;
  float s = 0.f, q = 0.f;
  for (long r = (long)blockIdx.x * 2 + sub; r < M; r += (long)gridDim.x * 2) {
    float v = x[r * 128 + c];
    s += v; q += v * v;
  }
  __shared__ float ls[256], lq[256];
  ls[threadIdx.x] = s; lq[threadIdx.x] = q;
  __syncthreads();
  if (sub == 0) {
    s += ls[c + 128]; q += lq[c + 128];
    atomicAdd(&sum[c], s); atomicAdd(&sq[c], q);
  }
}

__global__ void k_bnapply(float* x, const float* sum, const float* sq,
                          const float* g, const float* b, int M) {
  long i = (long)blockIdx.x * blockDim.x + threadIdx.x;
  if (i >= (long)M * 128) return;
  int c = i & 127;
  float mu = sum[c] / (float)M;
  float var = sq[c] / (float)M - mu * mu;
  x[i] = tanhf((x[i] - mu) * rsqrtf(var + BN_EPS) * g[c] + b[c]);
}

__global__ void k_poolmax(const float* gate, const int* batch, float* gmax) {
  long i = (long)blockIdx.x * blockDim.x + threadIdx.x;
  if (i >= (long)N_NODES * 128) return;
  int n = i >> 7, c = i & 127;
  atomicMaxF(&gmax[(long)batch[n] * 128 + c], gate[i]);
}

__global__ void k_poolex(float* gate, const int* batch, const float* gmax, float* denP) {
  long i = (long)blockIdx.x * blockDim.x + threadIdx.x;
  if (i >= (long)N_NODES * 128) return;
  int n = i >> 7, c = i & 127;
  float e = expf(gate[i] - gmax[(long)batch[n] * 128 + c]);
  gate[i] = e;
  atomicAdd(&denP[(long)batch[n] * 128 + c], e);
}

__global__ void k_poolacc(const float* gate, const int* batch, const float* denP,
                          const float* h, float* pooled) {
  long i = (long)blockIdx.x * blockDim.x + threadIdx.x;
  if (i >= (long)N_NODES * 128) return;
  int n = i >> 7, c = i & 127;
  long gidx = (long)batch[n] * 128 + c;
  float alpha = gate[i] / fmaxf(denP[gidx], 1e-16f);
  atomicAdd(&pooled[gidx], alpha * h[i]);
}

static inline int cdiv(long a, long b) { return (int)((a + b - 1) / b); }

extern "C" void kernel_launch(void* const* d_in, const int* in_sizes, int n_in,
                              void* d_out, int out_size, void* d_ws, size_t ws_size,
                              hipStream_t stream) {
  (void)in_sizes; (void)n_in; (void)out_size; (void)ws_size;
  const float* x     = (const float*)d_in[0];
  const int*   ei    = (const int*)d_in[1];
  const float* eattr = (const float*)d_in[2];
  const int*   batch = (const int*)d_in[3];
  const float *Wl1 = (const float*)d_in[4],  *bl1 = (const float*)d_in[5];
  const float *Wr1 = (const float*)d_in[6],  *br1 = (const float*)d_in[7];
  const float *We1 = (const float*)d_in[8],  *att1 = (const float*)d_in[9];
  // cb1 (d_in[10]) cancels in BatchNorm (per-column constant), skip.
  const float *Wl2 = (const float*)d_in[11], *bl2 = (const float*)d_in[12];
  const float *Wr2 = (const float*)d_in[13], *br2 = (const float*)d_in[14];
  const float *We2 = (const float*)d_in[15], *att2 = (const float*)d_in[16];
  const float *g1 = (const float*)d_in[18], *be1 = (const float*)d_in[19];
  const float *g2 = (const float*)d_in[20], *be2 = (const float*)d_in[21];
  const float *A1 = (const float*)d_in[22], *a1 = (const float*)d_in[23];
  const float *A2 = (const float*)d_in[24], *a2 = (const float*)d_in[25];
  const float *Wf = (const float*)d_in[26], *bf = (const float*)d_in[27];
  const int *src = ei, *dstp = ei + N_EDGES;
  float* out = (float*)d_out;

  // workspace carve-up (floats)
  float* EW     = (float*)d_ws;                        // E*128
  float* xl     = EW     + (long)N_EDGES * HDIM;       // N*128
  float* xr     = xl     + (long)N_NODES * HDIM;       // N*128
  float* h1     = xr     + (long)N_NODES * HDIM;       // N*128 (conv1 out / post-BN)
  float* h2     = h1     + (long)N_NODES * HDIM;       // N*128 (conv2 out / post-BN)
  float* loopEW = h2     + (long)N_NODES * HDIM;       // N*128
  float* deg    = loopEW + (long)N_NODES * HDIM;       // N
  float* score  = deg    + N_NODES;                    // E+N
  float* exv    = score  + (N_EDGES + N_NODES);        // E+N
  float* emax   = exv    + (N_EDGES + N_NODES);        // N
  float* den    = emax   + N_NODES;                    // N
  float* bns    = den    + N_NODES;                    // 128
  float* bnq    = bns    + HDIM;                       // 128
  float* gmax   = bnq    + HDIM;                       // G*128
  float* denP   = gmax   + (long)N_GRAPHS * HDIM;      // G*128
  float* pooled = denP   + (long)N_GRAPHS * HDIM;      // G*128

  const long NH = (long)N_NODES * HDIM;
  const long GH = (long)N_GRAPHS * HDIM;
  const int EN = N_EDGES + N_NODES;
  const int wave_blocks = cdiv((long)EN * 32, 256);
  const int edge_wave_blocks = cdiv((long)N_EDGES * 32, 256);

#define FILL(p, v, n) k_fill<<<cdiv((n), 256), 256, 0, stream>>>((p), (v), (long)(n))
#define GEMM(A_, W_, B_, C_, M_, act_) \
  k_gemm128<<<cdiv((M_), 64), 256, 0, stream>>>((A_), (W_), (B_), (C_), (M_), (act_))

  // ---------------- GATv2 layer 1 ----------------
  GEMM(x, Wl1, bl1, xl, N_NODES, 0);
  GEMM(x, Wr1, br1, xr, N_NODES, 0);
  GEMM(eattr, We1, nullptr, EW, N_EDGES, 0);
  FILL(deg, 0.f, N_NODES);
  FILL(loopEW, 0.f, NH);
  FILL(emax, -3.0e38f, N_NODES);
  FILL(den, 0.f, N_NODES);
  FILL(h1, 0.f, NH);
  k_scatter_loop<<<edge_wave_blocks, 256, 0, stream>>>(EW, dstp, deg, loopEW);
  k_loop_div<<<cdiv(NH, 256), 256, 0, stream>>>(loopEW, deg);
  k_score<<<wave_blocks, 256, 0, stream>>>(xl, xr, EW, loopEW, src, dstp, att1, score, emax);
  k_exden<<<cdiv(EN, 256), 256, 0, stream>>>(score, emax, dstp, exv, den);
  k_aggr<<<wave_blocks, 256, 0, stream>>>(exv, den, xl, src, dstp, h1);
  FILL(bns, 0.f, HDIM); FILL(bnq, 0.f, HDIM);
  k_bnstats<<<256, 256, 0, stream>>>(h1, bns, bnq, N_NODES);
  k_bnapply<<<cdiv(NH, 256), 256, 0, stream>>>(h1, bns, bnq, g1, be1, N_NODES);

  // ---------------- GATv2 layer 2 ----------------
  GEMM(h1, Wl2, bl2, xl, N_NODES, 0);
  GEMM(h1, Wr2, br2, xr, N_NODES, 0);
  GEMM(eattr, We2, nullptr, EW, N_EDGES, 0);
  FILL(deg, 0.f, N_NODES);
  FILL(loopEW, 0.f, NH);
  FILL(emax, -3.0e38f, N_NODES);
  FILL(den, 0.f, N_NODES);
  FILL(h2, 0.f, NH);
  k_scatter_loop<<<edge_wave_blocks, 256, 0, stream>>>(EW, dstp, deg, loopEW);
  k_loop_div<<<cdiv(NH, 256), 256, 0, stream>>>(loopEW, deg);
  k_score<<<wave_blocks, 256, 0, stream>>>(xl, xr, EW, loopEW, src, dstp, att2, score, emax);
  k_exden<<<cdiv(EN, 256), 256, 0, stream>>>(score, emax, dstp, exv, den);
  k_aggr<<<wave_blocks, 256, 0, stream>>>(exv, den, xl, src, dstp, h2);
  FILL(bns, 0.f, HDIM); FILL(bnq, 0.f, HDIM);
  k_bnstats<<<256, 256, 0, stream>>>(h2, bns, bnq, N_NODES);
  k_bnapply<<<cdiv(NH, 256), 256, 0, stream>>>(h2, bns, bnq, g2, be2, N_NODES);

  // ---------------- Attentional aggregation ----------------
  GEMM(h2, A1, a1, xl, N_NODES, 1);        // tanh(h@A1+a1)
  GEMM(xl, A2, a2, xr, N_NODES, 0);        // gate
  FILL(gmax, -3.0e38f, GH);
  FILL(denP, 0.f, GH);
  FILL(pooled, 0.f, GH);
  k_poolmax<<<cdiv(NH, 256), 256, 0, stream>>>(xr, batch, gmax);
  k_poolex<<<cdiv(NH, 256), 256, 0, stream>>>(xr, batch, gmax, denP);
  k_poolacc<<<cdiv(NH, 256), 256, 0, stream>>>(xr, batch, denP, h2, pooled);

  // ---------------- Final linear: axis = tanh(pooled@Wf+bf)*pi ----------------
  GEMM(pooled, Wf, bf, out, N_GRAPHS, 2);
  FILL(out + GH, 0.f, GH);   // aperture = zeros

#undef FILL
#undef GEMM
}
